// APTODE_55705725829573
// MI455X (gfx1250) — matrix operand order repaired and checked by reading, not compiled, not run on hardware
//
#include <hip/hip_runtime.h>
#include <math.h>

// ---------------------------------------------------------------------------
// APT-ODE on MI455X (gfx1250, wave32, WMMA).
//
// Roofline: ~1 GFLOP total, ~2MB of weights -> fully L2 resident. The ODE is a
// sequential Euler chain of small GEMVs; we batch the 4 sequences in lockstep
// (dt=0 padding) so every vf() evaluation becomes three skinny GEMMs
// (16x256x512, 16x512x512, 16x512x128) executed with v_wmma_f32_16x16x32_bf16
// by a single persistent 1024-thread workgroup (32 waves, one 16-wide N tile
// each). Weights are pre-converted to bf16 and pre-swizzled into the exact
// WMMA B-operand lane layout (1KB per 32x16 tile -> one contiguous 32B load
// per lane). The encoder MLP is a real WMMA GEMM. The detached partition /
// time-grid planning is ported to the device (double precision like numpy).
// ---------------------------------------------------------------------------

#define BATCH   4
#define SEQL    128
#define DIM     128
#define HID     512
#define DIN     256           // 2*DIM
#define W_WIN   5
#define ETA     0.5f
#define NSUB    2
#define GAPV    1e-6
#define MAXSEG  32
#define MAXSTEP 256
#define NST     (MAXSTEP + 1)
#define OUTW    (2 + 2 * SEQL * DIM)   // 32770 floats per batch row

typedef __bf16 v16bf __attribute__((ext_vector_type(16)));
typedef __bf16 v8bf  __attribute__((ext_vector_type(8)));
typedef float  v8f   __attribute__((ext_vector_type(8)));

// --------------------------- WMMA helpers ----------------------------------

__device__ inline v8f wmma_bf16(v16bf a, v16bf b, v8f c) {
    // emits v_wmma_f32_16x16x32_bf16
    return __builtin_amdgcn_wmma_f32_16x16x32_bf16(
        /*neg_a=*/false, a, /*neg_b=*/false, b,
        /*c_mod=*/(short)0, c, /*reuse_a=*/false, /*reuse_b=*/false);
}

// A-matrix (16xK bf16, row-major in LDS with leading dim `ld` halves).
// Per ISA 7.12.2: lane L<16 holds row L, K = {kb+h8..kb+h8+7, kb+16+h8..};
// h8 = (lane>=16)?8:0.  Two 16B LDS loads per fragment.
__device__ inline v16bf load_a_frag(const __bf16* A, int ld, int kbase, int lane) {
    int r = lane & 15;
    int h8 = (lane >> 4) << 3;
    const __bf16* p0 = A + r * ld + kbase + h8;
    const __bf16* p1 = p0 + 16;
    v8bf lo = *(const v8bf*)p0;
    v8bf hi = *(const v8bf*)p1;
    return __builtin_shufflevector(lo, hi, 0,1,2,3,4,5,6,7,8,9,10,11,12,13,14,15);
}

// B-matrix (32x16 bf16) from a pre-swizzled blob: tile (nt,kt) is 1KB,
// lane L's 16 halves are contiguous at L*32 bytes. One 32B load per lane.
__device__ inline v16bf load_b_frag(const __bf16* blob, int ktiles, int nt, int kt, int lane) {
    const __bf16* p = blob + ((size_t)(nt * ktiles + kt) << 9) + (lane << 4);
    return *(const v16bf*)p;
}

__device__ inline float softplusf(float x) {
    return (x > 20.f) ? x : log1pf(expf(x));
}

// ----------------------- kernel 1: weight prep -----------------------------
// fp32 (K,N) row-major -> bf16 WMMA-B swizzled blob.
__device__ inline void swizzle_elem(const float* src, __bf16* dst, int K, int N, int e) {
    int tile   = e >> 9;
    int within = e & 511;
    int lane   = within >> 4;
    int i      = within & 15;
    int ktiles = K >> 5;
    int nt = tile / ktiles;
    int kt = tile - nt * ktiles;
    int n = nt * 16 + (lane & 15);
    int k = kt * 32 + ((lane >> 4) << 3) + (i & 7) + ((i >> 3) << 4);
    dst[e] = (__bf16)src[k * N + n];
}

__global__ void prep_weights(const float* mw1, const float* mw2,
                             const float* vw1, const float* vw2, const float* vw3,
                             __bf16* mw1b, __bf16* mw2b,
                             __bf16* vw1b, __bf16* vw2b, __bf16* vw3b) {
    int t = blockIdx.x * blockDim.x + threadIdx.x;
    const int n1 = 128 * 128, n2 = 128 * 128, n3 = 256 * 512, n4 = 512 * 512, n5 = 512 * 128;
    if (t < n1) { swizzle_elem(mw1, mw1b, 128, 128, t); return; } t -= n1;
    if (t < n2) { swizzle_elem(mw2, mw2b, 128, 128, t); return; } t -= n2;
    if (t < n3) { swizzle_elem(vw1, vw1b, 256, 512, t); return; } t -= n3;
    if (t < n4) { swizzle_elem(vw2, vw2b, 512, 512, t); return; } t -= n4;
    if (t < n5) { swizzle_elem(vw3, vw3b, 512, 128, t); return; }
}

// ----------------------- kernel 2: encoder (WMMA) --------------------------
// enc = relu(emb[hi] @ W1 + b1) @ W2 + b2 ; also writes traj_t part of d_out.
__global__ __launch_bounds__(256) void encoder_kernel(
        const float* emb, const int* hi,
        const float* b1, const float* b2,
        const __bf16* w1b, const __bf16* w2b,
        float* enc, float* out) {
    __shared__ __align__(16) __bf16 A0[16][DIM];
    __shared__ __align__(16) __bf16 Hs[16][DIM];
    int rt   = blockIdx.x;           // 32 row tiles over B*L = 512 rows
    int tid  = threadIdx.x;
    int wave = tid >> 5, lane = tid & 31;

    for (int i = tid; i < 16 * DIM; i += 256) {
        int r = i >> 7, d = i & 127;
        int gr = rt * 16 + r;
        int item = hi[gr];                      // hi is (B*L,) flat
        A0[r][d] = (__bf16)emb[(size_t)item * DIM + d];
    }
    __syncthreads();

    int nt = wave;                              // 8 waves cover N = 128
    int col = lane & 15, half = lane >> 4;

    v8f acc = {0.f,0.f,0.f,0.f,0.f,0.f,0.f,0.f};
    for (int kt = 0; kt < 4; ++kt) {
        v16bf a = load_a_frag(&A0[0][0], DIM, kt * 32, lane);
        v16bf b = load_b_frag(w1b, 4, nt, kt, lane);
        acc = wmma_bf16(a, b, acc);
    }
    {
        int gc = nt * 16 + col;
        float bias = b1[gc];
        #pragma unroll
        for (int v = 0; v < 8; ++v) {
            int row = v + 8 * half;
            float x = acc[v] + bias;
            Hs[row][gc] = (__bf16)(x > 0.f ? x : 0.f);
        }
    }
    __syncthreads();

    v8f acc2 = {0.f,0.f,0.f,0.f,0.f,0.f,0.f,0.f};
    for (int kt = 0; kt < 4; ++kt) {
        v16bf a = load_a_frag(&Hs[0][0], DIM, kt * 32, lane);
        v16bf b = load_b_frag(w2b, 4, nt, kt, lane);
        acc2 = wmma_bf16(a, b, acc2);
    }
    {
        int gc = nt * 16 + col;
        float bias = b2[gc];
        #pragma unroll
        for (int v = 0; v < 8; ++v) {
            int row = v + 8 * half;
            int gr = rt * 16 + row;
            int b_ = gr >> 7, l = gr & 127;
            float val = acc2[v] + bias;
            enc[((size_t)(b_ * SEQL + l)) * DIM + gc] = val;
            out[(size_t)b_ * OUTW + 2 + SEQL * DIM + l * DIM + gc] = val;  // traj_t
        }
    }
}

// ----------------------- kernel 3: partition planning ----------------------

__device__ inline float blk_sum(float v, float* red, int t) {
    red[t] = v; __syncthreads();
    for (int o = 64; o > 0; o >>= 1) { if (t < o) red[t] += red[t + o]; __syncthreads(); }
    float r = red[0]; __syncthreads();
    return r;
}
__device__ inline float blk_max(float v, float* red, int t) {
    red[t] = v; __syncthreads();
    for (int o = 64; o > 0; o >>= 1) { if (t < o) red[t] = fmaxf(red[t], red[t + o]); __syncthreads(); }
    float r = red[0]; __syncthreads();
    return r;
}

__global__ __launch_bounds__(128) void plan_kernel(
        const float* enc, const float* ht, const int* hl, const float* pt,
        float* dtW, int* segidW, int* outidxW, float* envW,
        int* KstepsW, int* nsegW) {
    __shared__ float  red[128];
    __shared__ double sb[40];       // bounds
    __shared__ int    sstarts[40];
    __shared__ double sts[192];     // time grid of current segment
    __shared__ int    si_[10];      // 0:j 1:nb 2:ns 3:s 4:e 5:m 6:base 7:nint

    int b = blockIdx.x;
    int t = threadIdx.x;
    int n = hl[b];
    const float* encb  = enc + (size_t)b * SEQL * DIM;
    const float* times = ht + (size_t)b * SEQL;

    if (t == 0) {
        si_[0] = 0; si_[1] = 1; si_[2] = 1; si_[6] = 0;
        sb[0] = (double)times[0]; sstarts[0] = 0;
    }
    // ---- JSD sliding-window change-point scan ----
    while (true) {
        __syncthreads();
        int j = si_[0];
        if (j > n - 2 * W_WIN) break;
        float ml = 0.f, mr = 0.f;
        for (int i = 0; i < W_WIN; ++i) {
            ml += encb[(j + i) * DIM + t];
            mr += encb[(j + W_WIN + i) * DIM + t];
        }
        ml *= (1.f / W_WIN); mr *= (1.f / W_WIN);
        float mx = blk_max(ml, red, t);
        float el = expf(ml - mx);
        float pl = el / blk_sum(el, red, t);
        mx = blk_max(mr, red, t);
        float er = expf(mr - mx);
        float pr = er / blk_sum(er, red, t);
        float p = pl + 1e-8f; p /= blk_sum(p, red, t);
        float q = pr + 1e-8f; q /= blk_sum(q, red, t);
        float m = 0.5f * (p + q);
        float term = 0.5f * p * logf(p / m) + 0.5f * q * logf(q / m);
        float jsd = blk_sum(term, red, t);
        if (t == 0) {
            if (jsd > ETA && si_[1] < 34 && si_[2] < MAXSEG) {
                sb[si_[1]++] = 0.5 * ((double)times[j + W_WIN - 1] + (double)times[j + W_WIN]);
                sstarts[si_[2]++] = j + W_WIN;
                si_[0] = j + W_WIN;
            } else {
                si_[0] = j + 1;
            }
        }
    }
    __syncthreads();
    if (t == 0) {
        if ((double)times[n - 1] > sb[si_[1] - 1]) sb[si_[1]++] = (double)times[n - 1];
        nsegW[b] = si_[2];
    }
    __syncthreads();
    int nseg = si_[2];
    int nb   = si_[1];

    // ---- per-segment grid construction + plan emission ----
    for (int seg = 0; seg < nseg; ++seg) {
        __syncthreads();
        if (t == 0) {
            int s = sstarts[seg];
            int e = (seg + 1 < nseg) ? sstarts[seg + 1] - 1 : n - 1;
            double t0 = sb[seg];
            double t1 = (seg + 1 < nb) ? sb[seg + 1] : sb[nb - 1];
            int m = 0;
            sts[m++] = t0;
            for (int i = s; i <= e; ++i) {
                double ti = (double)times[i];
                if (ti >= t0 && ti <= t1 && ti > sts[m - 1]) sts[m++] = ti;
            }
            if (t1 > sts[m - 1]) sts[m++] = t1;
            for (int i = 1; i < m; ++i)
                if (sts[i] - sts[i - 1] < GAPV) sts[i] = sts[i - 1] + GAPV;
            int nint = m - 1;
            if (si_[6] + nint > MAXSTEP) nint = MAXSTEP - si_[6];
            si_[3] = s; si_[4] = e; si_[5] = m; si_[7] = nint;
        }
        __syncthreads();
        int s = si_[3], e = si_[4], m = si_[5], base = si_[6], nint = si_[7];
        // nearest-grid-point selection for each interaction time
        for (int i = s + t; i <= e; i += 128) {
            double ti = (double)times[i];
            int best = 0; double bd = 1e300;
            for (int g = 0; g < m; ++g) {
                double d = fabs(sts[g] - ti);
                if (d < bd) { bd = d; best = g; }
            }
            if (best > nint) best = nint;
            outidxW[b * SEQL + i] = base + best;
        }
        for (int k = t; k < nint; k += 128) {
            dtW[b * MAXSTEP + base + k]    = (float)(sts[k + 1] - sts[k]);
            segidW[b * MAXSTEP + base + k] = seg;
        }
        { // env = mean of enc rows s..e (thread t = dim)
            float a = 0.f;
            for (int i = s; i <= e; ++i) a += encb[i * DIM + t];
            envW[((size_t)b * MAXSEG + seg) * DIM + t] = a / (float)(e - s + 1);
        }
        __syncthreads();
        if (t == 0) si_[6] = base + nint;
    }
    __syncthreads();
    if (t == 0) {
        int base = si_[6];
        double tl = sb[nb - 1];
        double tp = (double)pt[b];
        if (tp > tl + 1e-6 && base < MAXSTEP) {
            double t1v = tp;
            if (t1v - tl < GAPV) t1v = tl + GAPV;
            dtW[b * MAXSTEP + base]    = (float)(t1v - tl);
            segidW[b * MAXSTEP + base] = nseg - 1;
            base++;
        }
        KstepsW[b] = base;
    }
}

// ----------------------- kernel 4: lockstep ODE (WMMA) ---------------------
// One persistent workgroup: 32 waves, one 16-wide N tile per wave.
__global__ __launch_bounds__(1024) void ode_kernel(
        const float* user_table, const int* u,
        const float* b1, const float* b2, const float* b3,
        const __bf16* w1b, const __bf16* w2b, const __bf16* w3b,
        const float* dtW, const int* segidW, const float* envW,
        const int* KstepsW, float* statesW) {
    __shared__ float zs[BATCH][DIM];
    __shared__ float fenv[BATCH][DIM];
    __shared__ __align__(16) __bf16 Xb[16][DIN];
    __shared__ __align__(16) __bf16 H1b[16][HID];
    __shared__ __align__(16) __bf16 H2b[16][HID];
    __shared__ float sc[BATCH];
    __shared__ int   shK[BATCH], shSid[BATCH], shKmax;

    int tid = threadIdx.x;
    int wave = tid >> 5, lane = tid & 31;
    int col = lane & 15, half = lane >> 4;

    if (tid == 0) {
        int mk = 0;
        for (int b = 0; b < BATCH; ++b) { int k = KstepsW[b]; shK[b] = k; if (k > mk) mk = k; }
        shKmax = mk;
        __builtin_prefetch(w2b, 0, 0);      // pull W2 toward L2/L0
    }
    if (tid < BATCH * DIM) {
        int b = tid >> 7, d = tid & 127;
        float zv = user_table[(size_t)u[b] * DIM + d];
        zs[b][d] = zv;
        statesW[((size_t)b * NST) * DIM + d] = zv;   // state 0
    }
    __syncthreads();
    int Kmax = shKmax;

    for (int k = 0; k < Kmax; ++k) {
        if (tid < BATCH) {
            int b = tid;
            float dt = 0.f; int sid = 0;
            if (k < shK[b]) { dt = dtW[b * MAXSTEP + k]; sid = segidW[b * MAXSTEP + k]; }
            sc[b] = dt / (float)NSUB;
            shSid[b] = sid;
        }
        __syncthreads();
        if (tid < BATCH * DIM) {
            int b = tid >> 7, d = tid & 127;
            fenv[b][d] = envW[((size_t)b * MAXSEG + shSid[b]) * DIM + d];
        }
        __syncthreads();

        for (int sub = 0; sub < NSUB; ++sub) {
            // Build A-operand X = [z | env] (rows 0..3), zero pad rows 4..15
            for (int i = tid; i < 16 * DIN; i += 1024) {
                int r = i >> 8, c = i & 255;
                float v = 0.f;
                if (r < BATCH) v = (c < DIM) ? zs[r][c] : fenv[r][c - DIM];
                Xb[r][c] = (__bf16)v;
            }
            __syncthreads();

            // Layer 1: 16x256 @ 256x512  (32 N tiles, 8 K tiles)
            {
                v8f acc = {0.f,0.f,0.f,0.f,0.f,0.f,0.f,0.f};
                int nt = wave;
                for (int kt = 0; kt < 8; ++kt) {
                    v16bf a = load_a_frag(&Xb[0][0], DIN, kt * 32, lane);
                    v16bf b = load_b_frag(w1b, 8, nt, kt, lane);
                    acc = wmma_bf16(a, b, acc);
                }
                int gc = nt * 16 + col;
                float bias = b1[gc];
                #pragma unroll
                for (int v = 0; v < 8; ++v) {
                    int row = v + 8 * half;
                    H1b[row][gc] = (__bf16)softplusf(acc[v] + bias);
                }
            }
            __syncthreads();

            // Layer 2: 16x512 @ 512x512  (32 N tiles, 16 K tiles)
            {
                v8f acc = {0.f,0.f,0.f,0.f,0.f,0.f,0.f,0.f};
                int nt = wave;
                for (int kt = 0; kt < 16; ++kt) {
                    v16bf a = load_a_frag(&H1b[0][0], HID, kt * 32, lane);
                    v16bf b = load_b_frag(w2b, 16, nt, kt, lane);
                    acc = wmma_bf16(a, b, acc);
                }
                int gc = nt * 16 + col;
                float bias = b2[gc];
                #pragma unroll
                for (int v = 0; v < 8; ++v) {
                    int row = v + 8 * half;
                    H2b[row][gc] = (__bf16)softplusf(acc[v] + bias);
                }
            }
            __syncthreads();

            // Layer 3: 16x512 @ 512x128  (8 N tiles on waves 0..7) + Euler update
            if (wave < 8) {
                v8f acc = {0.f,0.f,0.f,0.f,0.f,0.f,0.f,0.f};
                int nt = wave;
                for (int kt = 0; kt < 16; ++kt) {
                    v16bf a = load_a_frag(&H2b[0][0], HID, kt * 32, lane);
                    v16bf b = load_b_frag(w3b, 16, nt, kt, lane);
                    acc = wmma_bf16(a, b, acc);
                }
                int gc = nt * 16 + col;
                float bias = b3[gc];
                #pragma unroll
                for (int v = 0; v < 8; ++v) {
                    int row = v + 8 * half;
                    if (row < BATCH)
                        zs[row][gc] += sc[row] * (acc[v] + bias);
                }
            }
            __syncthreads();
        }
        if (tid < BATCH * DIM) {
            int b = tid >> 7, d = tid & 127;
            statesW[((size_t)b * NST + (k + 1)) * DIM + d] = zs[b][d];
        }
        __syncthreads();
    }
}

// ----------------------- kernel 5: finalize --------------------------------
__global__ __launch_bounds__(128) void finalize_kernel(
        const float* emb, const int* pos, const int* neg,
        const int* outidxW, const float* statesW, const int* KstepsW,
        float* out) {
    __shared__ float red[128];
    int b = blockIdx.x, t = threadIdx.x;
    int K = KstepsW[b];
    const float* zf = statesW + ((size_t)b * NST + K) * DIM;
    float zd = zf[t];
    float sp = blk_sum(zd * emb[(size_t)pos[b] * DIM + t], red, t);
    float sn = blk_sum(zd * emb[(size_t)neg[b] * DIM + t], red, t);
    if (t == 0) {
        out[(size_t)b * OUTW + 0] = sp;
        out[(size_t)b * OUTW + 1] = sn;
    }
    for (int i = t; i < SEQL * DIM; i += 128) {
        int l = i >> 7, d = i & 127;
        int g = outidxW[b * SEQL + l];
        out[(size_t)b * OUTW + 2 + i] = statesW[((size_t)b * NST + g) * DIM + d];
    }
}

// ----------------------- launch --------------------------------------------
extern "C" void kernel_launch(void* const* d_in, const int* in_sizes, int n_in,
                              void* d_out, int out_size, void* d_ws, size_t ws_size,
                              hipStream_t stream) {
    const float* emb   = (const float*)d_in[0];
    const float* mw1   = (const float*)d_in[1];
    const float* mb1   = (const float*)d_in[2];
    const float* mw2   = (const float*)d_in[3];
    const float* mb2   = (const float*)d_in[4];
    const float* usert = (const float*)d_in[5];
    const float* vw1   = (const float*)d_in[6];
    const float* vb1   = (const float*)d_in[7];
    const float* vw2   = (const float*)d_in[8];
    const float* vb2   = (const float*)d_in[9];
    const float* vw3   = (const float*)d_in[10];
    const float* vb3   = (const float*)d_in[11];
    const int*   u     = (const int*)d_in[12];
    const int*   hi    = (const int*)d_in[13];
    const float* ht    = (const float*)d_in[14];
    const int*   hl    = (const int*)d_in[15];
    const int*   pos   = (const int*)d_in[16];
    const int*   neg   = (const int*)d_in[17];
    const float* pt    = (const float*)d_in[18];
    float* out = (float*)d_out;

    char* w = (char*)d_ws;
    auto alloc = [&](size_t sz) { char* p = w; w += (sz + 255) & ~(size_t)255; return p; };
    float*  enc     = (float*)  alloc((size_t)BATCH * SEQL * DIM * 4);
    __bf16* mw1b    = (__bf16*) alloc((size_t)128 * 128 * 2);
    __bf16* mw2b    = (__bf16*) alloc((size_t)128 * 128 * 2);
    __bf16* vw1b    = (__bf16*) alloc((size_t)256 * 512 * 2);
    __bf16* vw2b    = (__bf16*) alloc((size_t)512 * 512 * 2);
    __bf16* vw3b    = (__bf16*) alloc((size_t)512 * 128 * 2);
    float*  dtW     = (float*)  alloc((size_t)BATCH * MAXSTEP * 4);
    int*    segidW  = (int*)    alloc((size_t)BATCH * MAXSTEP * 4);
    int*    outidxW = (int*)    alloc((size_t)BATCH * SEQL * 4);
    float*  envW    = (float*)  alloc((size_t)BATCH * MAXSEG * DIM * 4);
    int*    KstepsW = (int*)    alloc((size_t)BATCH * 4);
    int*    nsegW   = (int*)    alloc((size_t)BATCH * 4);
    float*  statesW = (float*)  alloc((size_t)BATCH * NST * DIM * 4);

    const int prep_total = 128 * 128 * 2 + 256 * 512 + 512 * 512 + 512 * 128;
    prep_weights<<<(prep_total + 255) / 256, 256, 0, stream>>>(
        mw1, mw2, vw1, vw2, vw3, mw1b, mw2b, vw1b, vw2b, vw3b);

    encoder_kernel<<<(BATCH * SEQL) / 16, 256, 0, stream>>>(
        emb, hi, mb1, mb2, mw1b, mw2b, enc, out);

    plan_kernel<<<BATCH, 128, 0, stream>>>(
        enc, ht, hl, pt, dtW, segidW, outidxW, envW, KstepsW, nsegW);

    ode_kernel<<<1, 1024, 0, stream>>>(
        usert, u, vb1, vb2, vb3, vw1b, vw2b, vw3b,
        dtW, segidW, envW, KstepsW, statesW);

    finalize_kernel<<<BATCH, 128, 0, stream>>>(
        emb, pos, neg, outidxW, statesW, KstepsW, out);
}
